// MEMO_15204184227858
// MI455X (gfx1250) — matrix operand order repaired and checked by reading, not compiled, not run on hardware
//
#include <hip/hip_runtime.h>
#include <hip/hip_bf16.h>
#include <math.h>

// ---------------------------------------------------------------------------
// Types for CDNA5 WMMA
// ---------------------------------------------------------------------------
typedef __attribute__((ext_vector_type(16))) __bf16 v16bf;
typedef __attribute__((ext_vector_type(8)))  float  v8f;

#define LOG2PI_F 1.8378770664093453f

// Epilogue kinds
#define EPI_NONE       0
#define EPI_BIAS       1
#define EPI_BIAS_TANH  2
#define EPI_DIST       3   // C = rown[row] + coln[col] - 2*acc
#define EPI_ADDC_TANH  4   // C = tanh(C_old + acc)

// ---------------------------------------------------------------------------
// WMMA GEMM:  C[M,N] = epilogue( A[M,K] * B[K,N] )  (fp32 in, bf16 WMMA, fp32 acc)
// Block tile 128x128, K-tile 32, double-buffered LDS (2 x 16 KB).
// 256 threads = 8 waves (wave32), waves 2x4, each wave owns a 64x32 sub-tile
// = 4x2 WMMA 16x16 accumulators (8 v_wmma per wave per K-tile).
// Global->reg loads for tile kt+1 are issued before the WMMA block of tile kt
// so VMEM latency overlaps matrix ops; regs are converted to bf16 and stored
// to the alternate LDS buffer after compute; one barrier per iteration.
// TRANSB=true means B is stored row-major as [N,K] (e.g. codebook).
// Requires: M%128==0, N%128==0, K%32==0.
// ---------------------------------------------------------------------------
template<int EPI, bool TRANSB>
__global__ __launch_bounds__(256)
void gemm_wmma_bf16(const float* __restrict__ A, const float* __restrict__ Bm,
                    const float* __restrict__ bias, float* __restrict__ C,
                    int M, int N, int Kd, int lda, int ldb, int ldc,
                    const float* __restrict__ rown, const float* __restrict__ coln)
{
    __shared__ __bf16 As[2][128][32];  // [buf][row][k]   2 x 8 KB
    __shared__ __bf16 Bs[2][128][32];  // [buf][col][k]   2 x 8 KB (B staged transposed)

    const int tid   = threadIdx.x;
    const int lane  = tid & 31;
    const int wave  = tid >> 5;      // 0..7
    const int waveM = wave >> 2;     // 0..1  -> 64-row slab
    const int waveN = wave & 3;      // 0..3  -> 32-col slab
    const int l16   = lane & 15;
    const int lhalf = lane >> 4;     // 0 or 1

    const int m0 = blockIdx.y * 128;
    const int n0 = blockIdx.x * 128;

    // staging assignments
    const int ra = tid >> 1;          // A row / TRANSB-B col  (0..127)
    const int ca = (tid & 1) * 16;    // k sub-block (0 or 16)
    const int kbrow = tid >> 3;       // !TRANSB: B row k (0..31)
    const int nb    = (tid & 7) * 16; // !TRANSB: n sub-block

    v8f acc[4][2];
#pragma unroll
    for (int i = 0; i < 4; ++i)
#pragma unroll
        for (int j = 0; j < 2; ++j) { v8f z = {}; acc[i][j] = z; }

    float ar[16], br[16];

    // global -> regs for K-tile kt (float4 / global_load_b128 path)
    auto gload = [&](int kt) {
        const int kk = kt * 32;
        {
            const float4* p = (const float4*)(A + (size_t)(m0 + ra) * lda + (kk + ca));
#pragma unroll
            for (int j = 0; j < 4; ++j) {
                const float4 v = p[j];
                ar[4*j+0] = v.x; ar[4*j+1] = v.y; ar[4*j+2] = v.z; ar[4*j+3] = v.w;
            }
        }
        if (TRANSB) {
            const float4* p = (const float4*)(Bm + (size_t)(n0 + ra) * ldb + (kk + ca));
#pragma unroll
            for (int j = 0; j < 4; ++j) {
                const float4 v = p[j];
                br[4*j+0] = v.x; br[4*j+1] = v.y; br[4*j+2] = v.z; br[4*j+3] = v.w;
            }
        } else {
            const float4* p = (const float4*)(Bm + (size_t)(kk + kbrow) * ldb + (n0 + nb));
#pragma unroll
            for (int j = 0; j < 4; ++j) {
                const float4 v = p[j];
                br[4*j+0] = v.x; br[4*j+1] = v.y; br[4*j+2] = v.z; br[4*j+3] = v.w;
            }
        }
    };

    // regs -> bf16 LDS buffer
    auto sstore = [&](int buf) {
#pragma unroll
        for (int i = 0; i < 16; ++i) As[buf][ra][ca + i] = (__bf16)ar[i];
        if (TRANSB) {
#pragma unroll
            for (int i = 0; i < 16; ++i) Bs[buf][ra][ca + i] = (__bf16)br[i];
        } else {
#pragma unroll
            for (int i = 0; i < 16; ++i) Bs[buf][nb + i][kbrow] = (__bf16)br[i];
        }
    };

    const int nk = Kd >> 5;

    // prologue: fill buffer 0
    gload(0);
    sstore(0);
    __syncthreads();

    for (int kt = 0; kt < nk; ++kt) {
        const int  cur     = kt & 1;
        const bool hasNext = (kt + 1) < nk;

        // issue next tile's global loads early (latency hidden under WMMAs)
        if (hasNext) {
            gload(kt + 1);
            // warm L2 for the tile after that
            const int kt2 = (kt + 2 < nk) ? (kt + 2) : (nk - 1);
            __builtin_prefetch(A + (size_t)(m0 + ra) * lda + (kt2 * 32 + ca), 0, 3);
        }

        // ---- build fragments per ISA 7.12.2 layouts ----
        // A 16x32 bf16: lanes 0-15 row M=l16 K={0..7,16..23}; lanes 16-31 K={8..15,24..31}
        v16bf afr[4];
#pragma unroll
        for (int tm = 0; tm < 4; ++tm) {
            const int row = waveM * 64 + tm * 16 + l16;
            const int kb  = lhalf * 8;
            v16bf a;
#pragma unroll
            for (int i = 0; i < 8; ++i) {
                a[i]     = As[cur][row][kb + i];
                a[i + 8] = As[cur][row][16 + kb + i];
            }
            afr[tm] = a;
        }
        // B 32x16 bf16: lanes 0-15 col N=l16 K={0..7,16..23}; lanes 16-31 K={8..15,24..31}
        v16bf bfr[2];
#pragma unroll
        for (int tn = 0; tn < 2; ++tn) {
            const int col = waveN * 32 + tn * 16 + l16;
            const int kb  = lhalf * 8;
            v16bf b;
#pragma unroll
            for (int i = 0; i < 8; ++i) {
                b[i]     = Bs[cur][col][kb + i];
                b[i + 8] = Bs[cur][col][16 + kb + i];
            }
            bfr[tn] = b;
        }
#pragma unroll
        for (int tm = 0; tm < 4; ++tm)
#pragma unroll
            for (int tn = 0; tn < 2; ++tn)
                acc[tm][tn] = __builtin_amdgcn_wmma_f32_16x16x32_bf16(
                    false, afr[tm], false, bfr[tn],
                    (short)0, acc[tm][tn], false, false);

        // store next tile into the alternate buffer
        if (hasNext) sstore(1 - cur);
        __syncthreads();
    }

    // ---- epilogue: C/D layout: vgpr r, lanes0-15 -> M=r, lanes16-31 -> M=8+r, N=l16
#pragma unroll
    for (int tm = 0; tm < 4; ++tm) {
#pragma unroll
        for (int tn = 0; tn < 2; ++tn) {
            const int col   = n0 + waveN * 32 + tn * 16 + l16;
            const int rbase = m0 + waveM * 64 + tm * 16 + lhalf * 8;
            float bv = 0.0f;
            if (EPI == EPI_BIAS || EPI == EPI_BIAS_TANH) bv = bias[col];
            float cn = 0.0f;
            if (EPI == EPI_DIST) cn = coln[col];
#pragma unroll
            for (int r = 0; r < 8; ++r) {
                const size_t idx = (size_t)(rbase + r) * ldc + col;
                const float v = acc[tm][tn][r];
                if (EPI == EPI_NONE)           C[idx] = v;
                else if (EPI == EPI_BIAS)      C[idx] = v + bv;
                else if (EPI == EPI_BIAS_TANH) C[idx] = tanhf(v + bv);
                else if (EPI == EPI_DIST)      C[idx] = rown[rbase + r] + cn - 2.0f * v;
                else if (EPI == EPI_ADDC_TANH) C[idx] = tanhf(C[idx] + v);
            }
        }
    }
}

// ---------------------------------------------------------------------------
// Row sum of squares: out[row] = sum_j X[row][j]^2
// ---------------------------------------------------------------------------
__global__ __launch_bounds__(256)
void row_sumsq(const float* __restrict__ X, float* __restrict__ out, int cols)
{
    const int row = blockIdx.x;
    const float* p = X + (size_t)row * cols;
    float s = 0.0f;
    for (int i = threadIdx.x; i < cols; i += 256) { const float v = p[i]; s += v * v; }
    __shared__ float sm[256];
    sm[threadIdx.x] = s; __syncthreads();
    for (int off = 128; off > 0; off >>= 1) {
        if (threadIdx.x < off) sm[threadIdx.x] += sm[threadIdx.x + off];
        __syncthreads();
    }
    if (threadIdx.x == 0) out[row] = sm[0];
}

// ---------------------------------------------------------------------------
// Row sum of squared differences: out[row] = sum_j (Aa[row][j]-Bb[row][j])^2
// ---------------------------------------------------------------------------
__global__ __launch_bounds__(256)
void row_diffsq(const float* __restrict__ Aa, const float* __restrict__ Bb,
                float* __restrict__ out, int cols)
{
    const int row = blockIdx.x;
    const float* pa = Aa + (size_t)row * cols;
    const float* pb = Bb + (size_t)row * cols;
    float s = 0.0f;
    for (int i = threadIdx.x; i < cols; i += 256) { const float d = pa[i] - pb[i]; s += d * d; }
    __shared__ float sm[256];
    sm[threadIdx.x] = s; __syncthreads();
    for (int off = 128; off > 0; off >>= 1) {
        if (threadIdx.x < off) sm[threadIdx.x] += sm[threadIdx.x + off];
        __syncthreads();
    }
    if (threadIdx.x == 0) out[row] = sm[0];
}

// ---------------------------------------------------------------------------
// Per-row argmin over dist[B,Kc] (first-index tie-break), gather codebook row
// into quantized[B,D], write float(proposal), and per-row VQ squared error.
// ---------------------------------------------------------------------------
__global__ __launch_bounds__(256)
void argmin_gather_vq(const float* __restrict__ dist, const float* __restrict__ codebook,
                      const float* __restrict__ enc_out, float* __restrict__ quantized,
                      float* __restrict__ prop_f, float* __restrict__ vrow,
                      int Kc, int D)
{
    const int b = blockIdx.x;
    const int t = threadIdx.x;
    const float* dr = dist + (size_t)b * Kc;

    float best = 3.402823466e38f;
    int   bi   = 0x7fffffff;
    for (int i = t; i < Kc; i += 256) {
        const float v = dr[i];
        if (v < best || (v == best && i < bi)) { best = v; bi = i; }
    }
    __shared__ float sv[256];
    __shared__ int   si[256];
    sv[t] = best; si[t] = bi; __syncthreads();
    for (int off = 128; off > 0; off >>= 1) {
        if (t < off) {
            const float ov = sv[t + off]; const int oi = si[t + off];
            if (ov < sv[t] || (ov == sv[t] && oi < si[t])) { sv[t] = ov; si[t] = oi; }
        }
        __syncthreads();
    }
    const int p = si[0];
    if (t == 0) prop_f[b] = (float)p;

    const float* crow = codebook + (size_t)p * D;
    const float* er   = enc_out  + (size_t)b * D;
    float* qr         = quantized + (size_t)b * D;
    float s = 0.0f;
    for (int i = t; i < D; i += 256) {
        const float q = crow[i];
        qr[i] = q;
        const float d = er[i] - q;
        s += d * d;
    }
    __syncthreads();
    sv[t] = s; __syncthreads();
    for (int off = 128; off > 0; off >>= 1) {
        if (t < off) sv[t] += sv[t + off];
        __syncthreads();
    }
    if (t == 0) vrow[b] = sv[0];
}

// ---------------------------------------------------------------------------
// total = mean(recon_err) + 2*mean(vq_err)  (vq_loss == commit_loss in fwd)
// Deterministic single-block reduction over the per-row partials.
// ---------------------------------------------------------------------------
__global__ __launch_bounds__(256)
void finalize_total(const float* __restrict__ rrow, const float* __restrict__ vrow,
                    int Brows, int D, float* __restrict__ total_ws,
                    float* __restrict__ total_out)
{
    const int t = threadIdx.x;
    float s1 = 0.0f, s2 = 0.0f;
    for (int i = t; i < Brows; i += 256) { s1 += rrow[i]; s2 += vrow[i]; }
    __shared__ float sa[256];
    __shared__ float sb[256];
    sa[t] = s1; sb[t] = s2; __syncthreads();
    for (int off = 128; off > 0; off >>= 1) {
        if (t < off) { sa[t] += sa[t + off]; sb[t] += sb[t + off]; }
        __syncthreads();
    }
    if (t == 0) {
        const float denom = (float)Brows * (float)D;
        const float tot = sa[0] / denom + 2.0f * (sb[0] / denom);
        *total_ws = tot;
        *total_out = tot;
    }
}

// ---------------------------------------------------------------------------
// Policy loss: per row b,
//   logp_j = -0.5*((A-mu)/std)^2 - log_std - 0.5*log(2pi)
//   loss_pi[b] = sum_j 1/(exp(logp_j)+0.1);  loss[b] = loss_pi[b]*total
// ---------------------------------------------------------------------------
__global__ __launch_bounds__(256)
void policy_loss(const float* __restrict__ Aact, const float* __restrict__ mu,
                 const float* __restrict__ log_std, const float* __restrict__ total_ws,
                 float* __restrict__ loss_out, float* __restrict__ loss_pi_out, int OUTd)
{
    const int b = blockIdx.x;
    const int j = threadIdx.x;
    float lp = 0.0f;
    if (j < OUTd) {
        const float ls  = log_std[j];
        const float std = expf(ls);
        const float z   = (Aact[(size_t)b * OUTd + j] - mu[(size_t)b * OUTd + j]) / std;
        const float logp = -0.5f * z * z - ls - 0.5f * LOG2PI_F;
        lp = 1.0f / (expf(logp) + 0.1f);
    }
    __shared__ float sm[256];
    sm[j] = lp; __syncthreads();
    for (int off = 128; off > 0; off >>= 1) {
        if (j < off) sm[j] += sm[j + off];
        __syncthreads();
    }
    if (j == 0) {
        const float s = sm[0];
        loss_pi_out[b] = s;
        loss_out[b]    = s * (*total_ws);
    }
}

// ---------------------------------------------------------------------------
// Launch helpers
// ---------------------------------------------------------------------------
static inline void launch_gemm(int epi, bool transb,
                               const float* A, const float* Bm, const float* bias,
                               float* C, int M, int N, int K,
                               int lda, int ldb, int ldc,
                               const float* rown, const float* coln, hipStream_t s)
{
    dim3 grid(N / 128, M / 128), block(256);
    if (!transb) {
        switch (epi) {
        case EPI_NONE:      gemm_wmma_bf16<EPI_NONE,      false><<<grid, block, 0, s>>>(A, Bm, bias, C, M, N, K, lda, ldb, ldc, rown, coln); break;
        case EPI_BIAS:      gemm_wmma_bf16<EPI_BIAS,      false><<<grid, block, 0, s>>>(A, Bm, bias, C, M, N, K, lda, ldb, ldc, rown, coln); break;
        case EPI_BIAS_TANH: gemm_wmma_bf16<EPI_BIAS_TANH, false><<<grid, block, 0, s>>>(A, Bm, bias, C, M, N, K, lda, ldb, ldc, rown, coln); break;
        case EPI_ADDC_TANH: gemm_wmma_bf16<EPI_ADDC_TANH, false><<<grid, block, 0, s>>>(A, Bm, bias, C, M, N, K, lda, ldb, ldc, rown, coln); break;
        default:            gemm_wmma_bf16<EPI_DIST,      false><<<grid, block, 0, s>>>(A, Bm, bias, C, M, N, K, lda, ldb, ldc, rown, coln); break;
        }
    } else {
        switch (epi) {
        case EPI_DIST:      gemm_wmma_bf16<EPI_DIST,      true ><<<grid, block, 0, s>>>(A, Bm, bias, C, M, N, K, lda, ldb, ldc, rown, coln); break;
        default:            gemm_wmma_bf16<EPI_NONE,      true ><<<grid, block, 0, s>>>(A, Bm, bias, C, M, N, K, lda, ldb, ldc, rown, coln); break;
        }
    }
}

extern "C" void kernel_launch(void* const* d_in, const int* in_sizes, int n_in,
                              void* d_out, int out_size, void* d_ws, size_t ws_size,
                              hipStream_t stream)
{
    (void)in_sizes; (void)n_in; (void)out_size; (void)ws_size;

    const int Bn = 32768, OBS = 1024, OUTd = 256, Kc = 512, HENC = 1024;

    const float* X        = (const float*)d_in[0];
    const float* Delta_X  = (const float*)d_in[1];
    const float* Aact     = (const float*)d_in[2];
    const float* enc_w1   = (const float*)d_in[3];
    const float* enc_b1   = (const float*)d_in[4];
    const float* enc_w2   = (const float*)d_in[5];
    const float* enc_b2   = (const float*)d_in[6];
    const float* prenet_w = (const float*)d_in[7];
    const float* prenet_b = (const float*)d_in[8];
    const float* codebook = (const float*)d_in[9];
    const float* postnet_w= (const float*)d_in[10];
    const float* postnet_b= (const float*)d_in[11];
    const float* dec_w1   = (const float*)d_in[12];
    const float* dec_b1   = (const float*)d_in[13];
    const float* dec_w2   = (const float*)d_in[14];
    const float* dec_b2   = (const float*)d_in[15];
    const float* dec_w3   = (const float*)d_in[16];
    const float* dec_b3   = (const float*)d_in[17];
    const float* act_w1   = (const float*)d_in[18];
    const float* act_b1   = (const float*)d_in[19];
    const float* act_w2   = (const float*)d_in[20];
    const float* act_b2   = (const float*)d_in[21];
    const float* act_w3   = (const float*)d_in[22];
    const float* act_b3   = (const float*)d_in[23];
    const float* log_std  = (const float*)d_in[24];

    // Output layout: [loss(B), loss_pi(B), X(B*OBS), proposal(B), total(1)]
    float* out      = (float*)d_out;
    float* loss     = out;
    float* loss_pi  = out + (size_t)Bn;
    float* Xout     = out + (size_t)2 * Bn;
    float* prop_f   = out + (size_t)2 * Bn + (size_t)Bn * OBS;
    float* totalp   = prop_f + Bn;

    // Workspace carving (floats)
    float* w = (float*)d_ws;
    size_t o = 0;
    float* t0 = w + o; o += (size_t)Bn * 512;   // h1
    float* t1 = w + o; o += (size_t)Bn * 1024;  // h2 -> reused as quantized
    float* t2 = w + o; o += (size_t)Bn * 1024;  // enc_out
    float* t3 = w + o; o += (size_t)Bn * 512;   // dist
    float* t4 = w + o; o += (size_t)Bn * 1024;  // d (=tanh(post_q)), later mu
    float* t5 = w + o; o += (size_t)Bn * 2048;  // dec h1, later actor h1
    float* t6 = w + o; o += (size_t)Bn * 2048;  // dec h2, later actor h2
    float* t7 = w + o; o += (size_t)Bn * 1024;  // recon
    float* xnorm = w + o; o += (size_t)Bn;
    float* enorm = w + o; o += (size_t)Kc;
    float* vrow  = w + o; o += (size_t)Bn;
    float* rrow  = w + o; o += (size_t)Bn;
    float* total_ws = w + o; o += 1;

    // ---- VQ encoder ----
    launch_gemm(EPI_BIAS_TANH, false, Delta_X, enc_w1, enc_b1, t0, Bn, 512, 1024, 1024, 512, 512, nullptr, nullptr, stream);
    launch_gemm(EPI_BIAS_TANH, false, t0, enc_w2, enc_b2, t1, Bn, 1024, 512, 512, 1024, 1024, nullptr, nullptr, stream);
    launch_gemm(EPI_BIAS,      false, t1, prenet_w, prenet_b, t2, Bn, 1024, 1024, 1024, 1024, 1024, nullptr, nullptr, stream);

    // ---- distances to codebook ----
    row_sumsq<<<Bn, 256, 0, stream>>>(t2, xnorm, OBS);
    row_sumsq<<<Kc, 256, 0, stream>>>(codebook, enorm, OBS);
    launch_gemm(EPI_DIST, true, t2, codebook, nullptr, t3, Bn, Kc, 1024, 1024, 1024, Kc, xnorm, enorm, stream);

    // ---- argmin + gather + per-row VQ error ----
    argmin_gather_vq<<<Bn, 256, 0, stream>>>(t3, codebook, t2, t1, prop_f, vrow, Kc, OBS);

    // ---- postnet (fused d = tanh(post_q)) + decoder ----
    launch_gemm(EPI_BIAS_TANH, false, t1, postnet_w, postnet_b, t4, Bn, HENC, 1024, 1024, HENC, HENC, nullptr, nullptr, stream);
    launch_gemm(EPI_BIAS_TANH, false, t4, dec_w1, dec_b1, t5, Bn, 2048, 1024, 1024, 2048, 2048, nullptr, nullptr, stream);
    launch_gemm(EPI_BIAS_TANH, false, t5, dec_w2, dec_b2, t6, Bn, 2048, 2048, 2048, 2048, 2048, nullptr, nullptr, stream);
    launch_gemm(EPI_BIAS,      false, t6, dec_w3, dec_b3, t7, Bn, 1024, 2048, 2048, 1024, 1024, nullptr, nullptr, stream);

    // ---- VQ criterion total ----
    row_diffsq<<<Bn, 256, 0, stream>>>(Delta_X, t7, rrow, OBS);
    finalize_total<<<1, 256, 0, stream>>>(rrow, vrow, Bn, OBS, total_ws, totalp);

    // ---- actor on [X, dist] (concat done as two GEMM passes) ----
    launch_gemm(EPI_BIAS,      false, X,  act_w1,              act_b1, t5, Bn, 2048, 1024, 1024, 2048, 2048, nullptr, nullptr, stream);
    launch_gemm(EPI_ADDC_TANH, false, t3, act_w1 + (size_t)1024 * 2048, nullptr, t5, Bn, 2048, 512, 512, 2048, 2048, nullptr, nullptr, stream);
    launch_gemm(EPI_BIAS_TANH, false, t5, act_w2, act_b2, t6, Bn, 2048, 2048, 2048, 2048, 2048, nullptr, nullptr, stream);
    launch_gemm(EPI_BIAS,      false, t6, act_w3, act_b3, t4, Bn, OUTd, 2048, 2048, OUTd, OUTd, nullptr, nullptr, stream);

    // ---- policy loss ----
    policy_loss<<<Bn, 256, 0, stream>>>(Aact, t4, log_std, total_ws, loss, loss_pi, OUTd);

    // ---- passthrough X output ----
    hipMemcpyAsync(Xout, X, (size_t)Bn * OBS * sizeof(float), hipMemcpyDeviceToDevice, stream);
}
